// MultiHeadAttention_65635690217864
// MI455X (gfx1250) — matrix operand order repaired and checked by reading, not compiled, not run on hardware
//
#include <hip/hip_runtime.h>
#include <hip/hip_bf16.h>

// ---------------------------------------------------------------------------
// Multi-head attention for MI455X (gfx1250), bf16 WMMA pipeline, software-
// pipelined K-loops (double-buffered fragments so loads overlap v_wmma).
//   B=4, S=2048, D_MODEL=1024, H=16, d_k=64
// Stage 1: weights fp32 -> bf16 (workspace)
// Stage 2: QKV projection GEMMs; Q,K stored [b][h][s][d] bf16, V stored
//          transposed [b][h][d][s] bf16; bias added in f32.
// Stage 3: flash attention, one wave per (b,h,16-query tile), 32-key chunks,
//          online softmax, P-tile relayout C->A via per-wave LDS.
// Stage 4: output projection GEMM, f32 output + bias.
// Workspace (72 MB): [0,8M) W bf16 | [8,24M) Qh | [24,40M) Kh | [40,56M) Vt
//                    | [56,72M) AttnOut bf16 [8192][1024]
// ---------------------------------------------------------------------------

#define BATCH   4
#define SEQ     2048
#define DMODEL  1024
#define NHEAD   16
#define DK      64

typedef __bf16        v16bf __attribute__((ext_vector_type(16)));
typedef float         v8f   __attribute__((ext_vector_type(8)));
typedef float         f32x4 __attribute__((ext_vector_type(4)));
typedef unsigned int  u32x4 __attribute__((ext_vector_type(4)));
typedef unsigned int  u32x2 __attribute__((ext_vector_type(2)));

union BFrag {                 // 32 bytes: one WMMA bf16 A- or B-fragment/lane
    v16bf          bf;
    u32x4          q[2];
    unsigned int   u[8];
    unsigned short s[16];
};

// ---- fp32 -> bf16 helpers -------------------------------------------------
// RNE version (cold paths: one-time tensor conversions / epilogues)
__device__ __forceinline__ unsigned f2bf_rne(float f) {
    union { float f; unsigned u; } x; x.f = f;
    unsigned r = x.u + 0x7FFFu + ((x.u >> 16) & 1u);
    return r >> 16;
}

#if defined(__has_builtin)
#if __has_builtin(__builtin_amdgcn_cvt_pk_bf16_f32)
#define HAVE_CVT_PK_BF16 1
#endif
#if __has_builtin(__builtin_amdgcn_perm)
#define HAVE_PERM 1
#endif
#endif

// Hot-path pack: two fp32 -> packed bf16x2 in ONE VALU op (v_perm_b32 grabs
// the high halves: result = {b[31:16], a[31:16]}). Truncation, <1 ulp.
__device__ __forceinline__ unsigned pk2_fast(float a, float b) {
#ifdef HAVE_CVT_PK_BF16
    auto t = __builtin_amdgcn_cvt_pk_bf16_f32(a, b);
    unsigned u; __builtin_memcpy(&u, &t, 4);
    return u;
#elif defined(HAVE_PERM)
    unsigned ua, ub;
    __builtin_memcpy(&ua, &a, 4);
    __builtin_memcpy(&ub, &b, 4);
    return __builtin_amdgcn_perm(ub, ua, 0x07060302u);  // {ub.b3,ub.b2,ua.b3,ua.b2}
#else
    unsigned ua, ub;
    __builtin_memcpy(&ua, &a, 4);
    __builtin_memcpy(&ub, &b, 4);
    return (ua >> 16) | (ub & 0xFFFF0000u);
#endif
}

// RNE pack for cold paths
__device__ __forceinline__ unsigned pk2_rne(float a, float b) {
#ifdef HAVE_CVT_PK_BF16
    auto t = __builtin_amdgcn_cvt_pk_bf16_f32(a, b);
    unsigned u; __builtin_memcpy(&u, &t, 4);
    return u;
#else
    return f2bf_rne(a) | (f2bf_rne(b) << 16);
#endif
}

__device__ __forceinline__ v8f wmma_bf16(v16bf a, v16bf b, v8f c) {
    return __builtin_amdgcn_wmma_f32_16x16x32_bf16(
        /*neg_a=*/false, a, /*neg_b=*/false, b,
        /*c_mod=*/(short)0, c, /*reuse_a=*/false, /*reuse_b=*/false);
}

// ---------------------------------------------------------------------------
// Stage 1: fp32 -> bf16 weight conversion. 4 matrices, 1 thread = 4 elements.
// ---------------------------------------------------------------------------
__global__ void cvt_weights_kernel(const float* __restrict__ Wq,
                                   const float* __restrict__ Wk,
                                   const float* __restrict__ Wv,
                                   const float* __restrict__ Wo,
                                   unsigned short* __restrict__ dst) {
    int idx = blockIdx.x * blockDim.x + threadIdx.x;     // 0 .. 2^20-1
    int mat = idx >> 18;                                 // 262144 float4 / mat
    int off = (idx & 262143) * 4;
    const float* src = (mat == 0) ? Wq : (mat == 1) ? Wk : (mat == 2) ? Wv : Wo;
    f32x4 vv = *(const f32x4*)(src + off);
    u32x2 r;
    r.x = pk2_rne(vv.x, vv.y);
    r.y = pk2_rne(vv.z, vv.w);
    *(u32x2*)(dst + mat * (DMODEL * DMODEL) + off) = r;
}

// ---------------------------------------------------------------------------
// Stage 2: projection GEMM, software-pipelined.
//   One wave = 16(M) x 64(N) strip; A raw fp32 double-buffered (packed to
//   bf16 with v_perm at consumption), 4 B bf16 fragments double-buffered.
// ---------------------------------------------------------------------------
__global__ void proj_kernel(const float* __restrict__ act_q,
                            const float* __restrict__ act_k,
                            const float* __restrict__ act_v,
                            const unsigned short* __restrict__ Wbf,
                            const float* __restrict__ bq,
                            const float* __restrict__ bk,
                            const float* __restrict__ bv,
                            unsigned short* __restrict__ Qh,
                            unsigned short* __restrict__ Kh,
                            unsigned short* __restrict__ Vt) {
    const int mode = blockIdx.z;                         // 0=Q 1=K 2=V
    const int wave = threadIdx.x >> 5;
    const int lane = threadIdx.x & 31;
    const int col  = lane & 15;
    const int half = lane >> 4;

    const int gw    = blockIdx.x * 8 + wave;             // 0..8191
    const int ntile = (gw & 15) * 64;                    // N base (64-wide)
    const int mbase = (gw >> 4) * 16;                    // M base (16 rows)

    const float* act  = (mode == 0) ? act_q : (mode == 1) ? act_k : act_v;
    const float* bias = (mode == 0) ? bq    : (mode == 1) ? bk    : bv;
    const unsigned short* W = Wbf + mode * (DMODEL * DMODEL);

    // per-lane base pointers with the half-wave K offsets baked in
    const float* actRow = act + (size_t)(mbase + col) * DMODEL + (half ? 16 : 0);
    const unsigned short* wp0 = W + (size_t)(ntile +  0 + col) * DMODEL + half * 16;
    const unsigned short* wp1 = W + (size_t)(ntile + 16 + col) * DMODEL + half * 16;
    const unsigned short* wp2 = W + (size_t)(ntile + 32 + col) * DMODEL + half * 16;
    const unsigned short* wp3 = W + (size_t)(ntile + 48 + col) * DMODEL + half * 16;

    f32x4 ar[2][4];
    BFrag br[2][4];

    // ---- prologue: issue loads for kk = 0 ----
    ar[0][0] = *(const f32x4*)(actRow + 0);
    ar[0][1] = *(const f32x4*)(actRow + 4);
    ar[0][2] = *(const f32x4*)(actRow + 8);
    ar[0][3] = *(const f32x4*)(actRow + 12);
    br[0][0].q[0] = *(const u32x4*)(wp0);  br[0][0].q[1] = *(const u32x4*)(wp0 + 8);
    br[0][1].q[0] = *(const u32x4*)(wp1);  br[0][1].q[1] = *(const u32x4*)(wp1 + 8);
    br[0][2].q[0] = *(const u32x4*)(wp2);  br[0][2].q[1] = *(const u32x4*)(wp2 + 8);
    br[0][3].q[0] = *(const u32x4*)(wp3);  br[0][3].q[1] = *(const u32x4*)(wp3 + 8);

    v8f acc0 = {}, acc1 = {}, acc2 = {}, acc3 = {};

    #pragma unroll 2
    for (int kk = 0; kk < DMODEL; kk += 32) {
        const int cb = (kk >> 5) & 1;
        const int nb = cb ^ 1;
        // ---- issue next iteration's loads first (overlap with wmma) ----
        if (kk + 32 < DMODEL) {
            const int kn = kk + 32;
            ar[nb][0] = *(const f32x4*)(actRow + kn);
            ar[nb][1] = *(const f32x4*)(actRow + kn + 4);
            ar[nb][2] = *(const f32x4*)(actRow + kn + 8);
            ar[nb][3] = *(const f32x4*)(actRow + kn + 12);
            __builtin_prefetch(actRow + kn + 32, 0, 0);
            br[nb][0].q[0] = *(const u32x4*)(wp0 + kn);  br[nb][0].q[1] = *(const u32x4*)(wp0 + kn + 8);
            br[nb][1].q[0] = *(const u32x4*)(wp1 + kn);  br[nb][1].q[1] = *(const u32x4*)(wp1 + kn + 8);
            br[nb][2].q[0] = *(const u32x4*)(wp2 + kn);  br[nb][2].q[1] = *(const u32x4*)(wp2 + kn + 8);
            br[nb][3].q[0] = *(const u32x4*)(wp3 + kn);  br[nb][3].q[1] = *(const u32x4*)(wp3 + kn + 8);
        }
        // ---- pack current A to bf16: 8x v_perm_b32, co-executes with XDL ----
        BFrag fa;
        fa.u[0] = pk2_fast(ar[cb][0].x, ar[cb][0].y); fa.u[1] = pk2_fast(ar[cb][0].z, ar[cb][0].w);
        fa.u[2] = pk2_fast(ar[cb][1].x, ar[cb][1].y); fa.u[3] = pk2_fast(ar[cb][1].z, ar[cb][1].w);
        fa.u[4] = pk2_fast(ar[cb][2].x, ar[cb][2].y); fa.u[5] = pk2_fast(ar[cb][2].z, ar[cb][2].w);
        fa.u[6] = pk2_fast(ar[cb][3].x, ar[cb][3].y); fa.u[7] = pk2_fast(ar[cb][3].z, ar[cb][3].w);

        acc0 = wmma_bf16(fa.bf, br[cb][0].bf, acc0);
        acc1 = wmma_bf16(fa.bf, br[cb][1].bf, acc1);
        acc2 = wmma_bf16(fa.bf, br[cb][2].bf, acc2);
        acc3 = wmma_bf16(fa.bf, br[cb][3].bf, acc3);
    }

    // ---- epilogue: +bias, bf16, scatter into head-split layouts ----
    const int rbase = half * 8;
    #pragma unroll
    for (int t = 0; t < 4; ++t) {
        const int gn = ntile + t * 16 + col;             // d_model column
        const float bi = bias[gn];
        const int h = gn >> 6, d = gn & 63;
        v8f acc = (t == 0) ? acc0 : (t == 1) ? acc1 : (t == 2) ? acc2 : acc3;
        #pragma unroll
        for (int v = 0; v < 8; ++v) {
            const int m = mbase + rbase + v;             // global token
            const int b = m >> 11, s = m & 2047;
            const unsigned short hv = (unsigned short)f2bf_rne(acc[v] + bi);
            if (mode < 2) {
                unsigned short* dst = (mode == 0) ? Qh : Kh;
                dst[((size_t)(b * NHEAD + h) * SEQ + s) * DK + d] = hv;
            } else {
                Vt[((size_t)(b * NHEAD + h) * DK + d) * SEQ + s] = hv;
            }
        }
    }
}

// ---------------------------------------------------------------------------
// Stage 3: flash attention, K-chunk fragments double-buffered.
// ---------------------------------------------------------------------------
__global__ void attn_kernel(const unsigned short* __restrict__ Qh,
                            const unsigned short* __restrict__ Kh,
                            const unsigned short* __restrict__ Vt,
                            unsigned short* __restrict__ Ao) {
    __shared__ unsigned short pT[8 * 16 * 32];           // 1KB per wave

    const int wave = threadIdx.x >> 5;
    const int lane = threadIdx.x & 31;
    const int col  = lane & 15;
    const int half = lane >> 4;
    const int pbase = wave * (16 * 32);

    const int gw = blockIdx.x * 8 + wave;                // 0..8191
    const int qt = gw & 127;                             // query tile
    const int bh = gw >> 7;                              // (b*16 + h)
    const int b  = bh >> 4;
    const int h  = bh & 15;

    const size_t headQK = (size_t)bh * SEQ * DK;
    const size_t headV  = (size_t)bh * DK * SEQ;

    // ---- Q A-fragments (16 x 64, two K-halves of d_k) ----
    const unsigned short* qRow = Qh + headQK + (size_t)(qt * 16 + col) * DK;
    const int ka = half * 16;
    BFrag qa0, qa1;
    qa0.q[0] = *(const u32x4*)(qRow + ka);
    qa0.q[1] = *(const u32x4*)(qRow + ka + 8);
    qa1.q[0] = *(const u32x4*)(qRow + 32 + ka);
    qa1.q[1] = *(const u32x4*)(qRow + 32 + ka + 8);

    // per-lane base pointers (half-wave K offsets baked in)
    const unsigned short* kBase = Kh + headQK + (size_t)col * DK + half * 16;
    const unsigned short* vBase = Vt + headV + (size_t)col * SEQ + half * 16;

    v8f o0 = {}, o1 = {}, o2 = {}, o3 = {};
    float rm[8], rs[8];
    #pragma unroll
    for (int v = 0; v < 8; ++v) { rm[v] = -1e30f; rs[v] = 0.0f; }

    const float scale = 0.125f;                          // 1/sqrt(64)

    // kb[buf][frag]: frag 0/1 = keys j..j+15 (d lo/hi), 2/3 = keys j+16..j+31
    BFrag kb[2][4];
    {
        const unsigned short* kp = kBase;
        kb[0][0].q[0] = *(const u32x4*)(kp);           kb[0][0].q[1] = *(const u32x4*)(kp + 8);
        kb[0][1].q[0] = *(const u32x4*)(kp + 32);      kb[0][1].q[1] = *(const u32x4*)(kp + 40);
        kb[0][2].q[0] = *(const u32x4*)(kp + 1024);    kb[0][2].q[1] = *(const u32x4*)(kp + 1032);
        kb[0][3].q[0] = *(const u32x4*)(kp + 1056);    kb[0][3].q[1] = *(const u32x4*)(kp + 1064);
    }

    #pragma unroll 2
    for (int j = 0; j < SEQ; j += 32) {                  // 64 key chunks
        const int cb = (j >> 5) & 1;
        const int nb = cb ^ 1;

        // ---- issue next chunk's K loads ----
        if (j + 32 < SEQ) {
            const unsigned short* kp = kBase + (size_t)(j + 32) * DK;
            kb[nb][0].q[0] = *(const u32x4*)(kp);           kb[nb][0].q[1] = *(const u32x4*)(kp + 8);
            kb[nb][1].q[0] = *(const u32x4*)(kp + 32);      kb[nb][1].q[1] = *(const u32x4*)(kp + 40);
            kb[nb][2].q[0] = *(const u32x4*)(kp + 1024);    kb[nb][2].q[1] = *(const u32x4*)(kp + 1032);
            kb[nb][3].q[0] = *(const u32x4*)(kp + 1056);    kb[nb][3].q[1] = *(const u32x4*)(kp + 1064);
        }
        // ---- issue this chunk's V loads (consumed after softmax) ----
        BFrag vb0, vb1, vb2, vb3;
        {
            const unsigned short* vp = vBase + j;
            vb0.q[0] = *(const u32x4*)(vp);                  vb0.q[1] = *(const u32x4*)(vp + 8);
            vb1.q[0] = *(const u32x4*)(vp + 16 * SEQ);       vb1.q[1] = *(const u32x4*)(vp + 16 * SEQ + 8);
            vb2.q[0] = *(const u32x4*)(vp + 32 * SEQ);       vb2.q[1] = *(const u32x4*)(vp + 32 * SEQ + 8);
            vb3.q[0] = *(const u32x4*)(vp + 48 * SEQ);       vb3.q[1] = *(const u32x4*)(vp + 48 * SEQ + 8);
        }

        // ---- scores: two 16x16 tiles, contract over d_k = 64 ----
        v8f s0 = {}, s1 = {};
        s0 = wmma_bf16(qa0.bf, kb[cb][0].bf, s0);
        s0 = wmma_bf16(qa1.bf, kb[cb][1].bf, s0);
        s1 = wmma_bf16(qa0.bf, kb[cb][2].bf, s1);
        s1 = wmma_bf16(qa1.bf, kb[cb][3].bf, s1);

        // ---- online softmax ----
        float corr[8], p0a[8], p1a[8];
        #pragma unroll
        for (int v = 0; v < 8; ++v) {
            float x0 = s0[v] * scale;
            float x1 = s1[v] * scale;
            float cm = fmaxf(x0, x1);
            #pragma unroll
            for (int msk = 1; msk < 16; msk <<= 1)
                cm = fmaxf(cm, __shfl_xor(cm, msk, 32));
            const float mnew = fmaxf(rm[v], cm);
            corr[v] = __expf(rm[v] - mnew);
            rm[v] = mnew;
            p0a[v] = __expf(x0 - mnew);
            p1a[v] = __expf(x1 - mnew);
            float part = p0a[v] + p1a[v];
            #pragma unroll
            for (int msk = 1; msk < 16; msk <<= 1)
                part += __shfl_xor(part, msk, 32);
            rs[v] = rs[v] * corr[v] + part;
            o0[v] *= corr[v]; o1[v] *= corr[v];
            o2[v] *= corr[v]; o3[v] *= corr[v];
        }

        // ---- P tile: C-layout f32 -> LDS -> A-layout bf16 ----
        // (truncating bf16 pack: p in [0,1], 1 shift per element)
        const int rb = half * 8;
        #pragma unroll
        for (int v = 0; v < 8; ++v) {
            unsigned u0, u1;
            __builtin_memcpy(&u0, &p0a[v], 4);
            __builtin_memcpy(&u1, &p1a[v], 4);
            pT[pbase + (rb + v) * 32 + col]      = (unsigned short)(u0 >> 16);
            pT[pbase + (rb + v) * 32 + 16 + col] = (unsigned short)(u1 >> 16);
        }
        BFrag pf;
        pf.q[0] = *(const u32x4*)&pT[pbase + col * 32 + ka];
        pf.q[1] = *(const u32x4*)&pT[pbase + col * 32 + ka + 8];

        // ---- O += P (16x32) x V (32x64) ----
        o0 = wmma_bf16(pf.bf, vb0.bf, o0);
        o1 = wmma_bf16(pf.bf, vb1.bf, o1);
        o2 = wmma_bf16(pf.bf, vb2.bf, o2);
        o3 = wmma_bf16(pf.bf, vb3.bf, o3);
    }

    // ---- normalize + store bf16 into [token][d_model] layout ----
    const int rb = half * 8;
    #pragma unroll
    for (int v = 0; v < 8; ++v) {
        const float inv = 1.0f / rs[v];
        const int s = qt * 16 + rb + v;
        const size_t row = ((size_t)b * SEQ + s) * DMODEL + h * DK;
        Ao[row + 0 * 16 + col] = (unsigned short)f2bf_rne(o0[v] * inv);
        Ao[row + 1 * 16 + col] = (unsigned short)f2bf_rne(o1[v] * inv);
        Ao[row + 2 * 16 + col] = (unsigned short)f2bf_rne(o2[v] * inv);
        Ao[row + 3 * 16 + col] = (unsigned short)f2bf_rne(o3[v] * inv);
    }
}

// ---------------------------------------------------------------------------
// Stage 4: output projection, software-pipelined. f32 out + bias.
// ---------------------------------------------------------------------------
__global__ void outproj_kernel(const unsigned short* __restrict__ Ao,
                               const unsigned short* __restrict__ Wo,
                               const float* __restrict__ bo,
                               float* __restrict__ out) {
    const int wave = threadIdx.x >> 5;
    const int lane = threadIdx.x & 31;
    const int col  = lane & 15;
    const int half = lane >> 4;

    const int gw    = blockIdx.x * 8 + wave;
    const int ntile = (gw & 15) * 64;
    const int mbase = (gw >> 4) * 16;

    const unsigned short* aRow = Ao + (size_t)(mbase + col) * DMODEL + (half ? 16 : 0);
    const unsigned short* wp0 = Wo + (size_t)(ntile +  0 + col) * DMODEL + half * 16;
    const unsigned short* wp1 = Wo + (size_t)(ntile + 16 + col) * DMODEL + half * 16;
    const unsigned short* wp2 = Wo + (size_t)(ntile + 32 + col) * DMODEL + half * 16;
    const unsigned short* wp3 = Wo + (size_t)(ntile + 48 + col) * DMODEL + half * 16;

    BFrag fa[2], br[2][4];
    fa[0].q[0] = *(const u32x4*)(aRow);
    fa[0].q[1] = *(const u32x4*)(aRow + 8);
    br[0][0].q[0] = *(const u32x4*)(wp0);  br[0][0].q[1] = *(const u32x4*)(wp0 + 8);
    br[0][1].q[0] = *(const u32x4*)(wp1);  br[0][1].q[1] = *(const u32x4*)(wp1 + 8);
    br[0][2].q[0] = *(const u32x4*)(wp2);  br[0][2].q[1] = *(const u32x4*)(wp2 + 8);
    br[0][3].q[0] = *(const u32x4*)(wp3);  br[0][3].q[1] = *(const u32x4*)(wp3 + 8);

    v8f acc0 = {}, acc1 = {}, acc2 = {}, acc3 = {};

    #pragma unroll 2
    for (int kk = 0; kk < DMODEL; kk += 32) {
        const int cb = (kk >> 5) & 1;
        const int nb = cb ^ 1;
        if (kk + 32 < DMODEL) {
            const int kn = kk + 32;
            fa[nb].q[0] = *(const u32x4*)(aRow + kn);
            fa[nb].q[1] = *(const u32x4*)(aRow + kn + 8);
            __builtin_prefetch(aRow + kn + 32, 0, 0);
            br[nb][0].q[0] = *(const u32x4*)(wp0 + kn);  br[nb][0].q[1] = *(const u32x4*)(wp0 + kn + 8);
            br[nb][1].q[0] = *(const u32x4*)(wp1 + kn);  br[nb][1].q[1] = *(const u32x4*)(wp1 + kn + 8);
            br[nb][2].q[0] = *(const u32x4*)(wp2 + kn);  br[nb][2].q[1] = *(const u32x4*)(wp2 + kn + 8);
            br[nb][3].q[0] = *(const u32x4*)(wp3 + kn);  br[nb][3].q[1] = *(const u32x4*)(wp3 + kn + 8);
        }
        acc0 = wmma_bf16(fa[cb].bf, br[cb][0].bf, acc0);
        acc1 = wmma_bf16(fa[cb].bf, br[cb][1].bf, acc1);
        acc2 = wmma_bf16(fa[cb].bf, br[cb][2].bf, acc2);
        acc3 = wmma_bf16(fa[cb].bf, br[cb][3].bf, acc3);
    }

    const int rbase = half * 8;
    #pragma unroll
    for (int t = 0; t < 4; ++t) {
        const int gn = ntile + t * 16 + col;
        const float bi = bo[gn];
        v8f acc = (t == 0) ? acc0 : (t == 1) ? acc1 : (t == 2) ? acc2 : acc3;
        #pragma unroll
        for (int v = 0; v < 8; ++v) {
            const int m = mbase + rbase + v;
            out[(size_t)m * DMODEL + gn] = acc[v] + bi;
        }
    }
}

// ---------------------------------------------------------------------------
extern "C" void kernel_launch(void* const* d_in, const int* in_sizes, int n_in,
                              void* d_out, int out_size, void* d_ws, size_t ws_size,
                              hipStream_t stream) {
    (void)in_sizes; (void)n_in; (void)out_size; (void)ws_size;

    const float* q  = (const float*)d_in[0];
    const float* k  = (const float*)d_in[1];
    const float* v  = (const float*)d_in[2];
    const float* Wq = (const float*)d_in[3];
    const float* bq = (const float*)d_in[4];
    const float* Wk = (const float*)d_in[5];
    const float* bk = (const float*)d_in[6];
    const float* Wv = (const float*)d_in[7];
    const float* bv = (const float*)d_in[8];
    const float* Wo = (const float*)d_in[9];
    const float* bo = (const float*)d_in[10];

    const size_t MB = 1024 * 1024;
    unsigned short* wbf = (unsigned short*)d_ws;                        //  8 MB
    unsigned short* Qh  = (unsigned short*)((char*)d_ws +  8 * MB);     // 16 MB
    unsigned short* Kh  = (unsigned short*)((char*)d_ws + 24 * MB);     // 16 MB
    unsigned short* Vt  = (unsigned short*)((char*)d_ws + 40 * MB);     // 16 MB
    unsigned short* Ao  = (unsigned short*)((char*)d_ws + 56 * MB);     // 16 MB

    cvt_weights_kernel<<<4096, 256, 0, stream>>>(Wq, Wk, Wv, Wo, wbf);

    proj_kernel<<<dim3(1024, 1, 3), 256, 0, stream>>>(
        q, k, v, wbf, bq, bk, bv, Qh, Kh, Vt);

    attn_kernel<<<1024, 256, 0, stream>>>(Qh, Kh, Vt, Ao);

    outproj_kernel<<<1024, 256, 0, stream>>>(
        Ao, wbf + 3 * (size_t)(DMODEL * DMODEL), bo, (float*)d_out);
}
